// TinyGPT_33414845563559
// MI455X (gfx1250) — matrix operand order repaired and checked by reading, not compiled, not run on hardware
//
#include <hip/hip_runtime.h>

// ---------------- model constants ----------------
#define VOCAB   32000
#define BATCH   2
#define SEQ     2048
#define DMODEL  768
#define NHEAD   12
#define HEADD   64
#define NLAYER  4
#define FFDIM   3072
#define BT      (BATCH * SEQ)      // 4096
#define QKVLD   (3 * DMODEL)       // 2304

typedef __attribute__((ext_vector_type(16))) _Float16     v16h;
typedef __attribute__((ext_vector_type(8)))  float        v8f;
typedef __attribute__((ext_vector_type(4)))  unsigned int u32x4;

union Frag32B { u32x4 u[2]; v16h v; };

// A-operand (16x32 f16) per ISA table: lane<16 holds row m=lane, K = k0+[0,8) and k0+[16,24);
// lane>=16 holds row m=lane-16, K = k0+[8,16) and k0+[24,32).
__device__ __forceinline__ v16h load_a_frag(const _Float16* base, int ld, int row0, int k0, int lane) {
    int l16 = lane & 15, g = lane >> 4;
    const u32x4* p = (const u32x4*)(base + (size_t)(row0 + l16) * ld + k0 + g * 8);
    Frag32B f;
    f.u[0] = p[0];   // 8 halves
    f.u[1] = p[2];   // +16 halves
    return f.v;
}

// B-operand (32x16 f16), from Bt stored [N,K] row-major: lane<16 -> col n=lane holds K = k0+[0,8) and k0+[16,24);
// lane>=16 -> col n=lane-16 holds K = k0+[8,16) and k0+[24,32). (matches A-operand table order)
__device__ __forceinline__ v16h load_b_frag(const _Float16* base, int ld, int n0, int k0, int lane) {
    int l16 = lane & 15, g = lane >> 4;
    const u32x4* p = (const u32x4*)(base + (size_t)(n0 + l16) * ld + k0 + g * 8);
    Frag32B f;
    f.u[0] = p[0];
    f.u[1] = p[2];
    return f.v;
}

__device__ __forceinline__ v8f wmma_f16(v16h a, v16h b, v8f c) {
    return __builtin_amdgcn_wmma_f32_16x16x32_f16(false, a, false, b, (short)0, c, false, false);
}

__device__ __forceinline__ float gelu_exact(float v) {
    return 0.5f * v * (1.0f + erff(v * 0.7071067811865476f));
}

// ---------------- elementwise helpers ----------------
__global__ void __launch_bounds__(256)
cvt_f16_kernel(const float* __restrict__ in, _Float16* __restrict__ out, long n) {
    long tid = (long)blockIdx.x * 256 + threadIdx.x;
    if (tid < n) out[tid] = (_Float16)in[tid];
}

// in: [K,N] f32 row-major -> out: [N,K] f16 row-major
__global__ void __launch_bounds__(256)
transpose_cvt_kernel(const float* __restrict__ in, _Float16* __restrict__ out, int K, int N) {
    long tid = (long)blockIdx.x * 256 + threadIdx.x;
    long total = (long)K * N;
    if (tid >= total) return;
    int k = (int)(tid / N);
    int n = (int)(tid % N);
    out[(size_t)n * K + k] = (_Float16)in[tid];
}

__global__ void __launch_bounds__(256)
embed_kernel(const int* __restrict__ idx, const float* __restrict__ tok,
             const float* __restrict__ pos, float* __restrict__ x) {
    long tid = (long)blockIdx.x * 256 + threadIdx.x;
    long total = (long)BT * DMODEL;
    if (tid >= total) return;
    int d = (int)(tid % DMODEL);
    long row = tid / DMODEL;           // b*T + t
    int t = (int)(row % SEQ);
    x[tid] = tok[(size_t)idx[row] * DMODEL + d] + pos[(size_t)t * DMODEL + d];
}

// Extract V from qkv (f16) and transpose per head: vt[(b,h)][hd][t]
__global__ void __launch_bounds__(256)
vtrans_kernel(const _Float16* __restrict__ qkv, _Float16* __restrict__ vt) {
    long tid = (long)blockIdx.x * 256 + threadIdx.x;
    long total = (long)BATCH * NHEAD * SEQ * HEADD;
    if (tid >= total) return;
    int hd = (int)(tid % HEADD);
    int h  = (int)((tid / HEADD) % NHEAD);
    int t  = (int)((tid / (HEADD * NHEAD)) % SEQ);
    int b  = (int)(tid / ((long)HEADD * NHEAD * SEQ));
    _Float16 val = qkv[((size_t)(b * SEQ + t)) * QKVLD + 2 * DMODEL + h * HEADD + hd];
    vt[(((size_t)(b * NHEAD + h)) * HEADD + hd) * SEQ + t] = val;
}

// ---------------- layernorm (row per block) ----------------
__global__ void __launch_bounds__(256)
layernorm_kernel(const float* __restrict__ x, const float* __restrict__ w,
                 const float* __restrict__ bsh, _Float16* __restrict__ out) {
    __shared__ float red[256];
    int row = blockIdx.x;
    const float* xr = x + (size_t)row * DMODEL;
    float s = 0.f;
    for (int i = threadIdx.x; i < DMODEL; i += 256) s += xr[i];
    red[threadIdx.x] = s; __syncthreads();
    for (int off = 128; off > 0; off >>= 1) {
        if (threadIdx.x < off) red[threadIdx.x] += red[threadIdx.x + off];
        __syncthreads();
    }
    float mu = red[0] * (1.0f / DMODEL);
    __syncthreads();
    float v = 0.f;
    for (int i = threadIdx.x; i < DMODEL; i += 256) { float d = xr[i] - mu; v += d * d; }
    red[threadIdx.x] = v; __syncthreads();
    for (int off = 128; off > 0; off >>= 1) {
        if (threadIdx.x < off) red[threadIdx.x] += red[threadIdx.x + off];
        __syncthreads();
    }
    float rinv = rsqrtf(red[0] * (1.0f / DMODEL) + 1e-5f);
    for (int i = threadIdx.x; i < DMODEL; i += 256)
        out[(size_t)row * DMODEL + i] = (_Float16)((xr[i] - mu) * rinv * w[i] + bsh[i]);
}

// ---------------- generic WMMA GEMM ----------------
// C[M,N] = A[M,K] (f16, ld=K) @ Bt[N,K]^T (f16, ld=K) [+bias][gelu][+resid]
// block = 128 threads (4 waves, 2x2), block tile 64(M) x 128(N); wave tile 32x64.
__global__ void __launch_bounds__(128)
gemm_f16_kernel(const _Float16* __restrict__ A, const _Float16* __restrict__ Bt,
                const float* __restrict__ bias, const float* __restrict__ resid,
                float* __restrict__ Cf, _Float16* __restrict__ Cb,
                int M, int N, int K, int act) {
    int lane = threadIdx.x & 31;
    int wave = threadIdx.x >> 5;
    int m0 = blockIdx.y * 64 + (wave >> 1) * 32;
    int n0 = blockIdx.x * 128 + (wave & 1) * 64;
    if (m0 >= M || n0 >= N) return;

    v8f acc[2][4];
    #pragma unroll
    for (int mi = 0; mi < 2; ++mi)
        #pragma unroll
        for (int j = 0; j < 4; ++j)
            #pragma unroll
            for (int q = 0; q < 8; ++q) acc[mi][j][q] = 0.f;

    for (int k0 = 0; k0 < K; k0 += 32) {
        v16h a0 = load_a_frag(A, K, m0,      k0, lane);
        v16h a1 = load_a_frag(A, K, m0 + 16, k0, lane);
        #pragma unroll
        for (int j = 0; j < 4; ++j) {
            v16h bf = load_b_frag(Bt, K, n0 + j * 16, k0, lane);
            acc[0][j] = wmma_f16(a0, bf, acc[0][j]);
            acc[1][j] = wmma_f16(a1, bf, acc[1][j]);
        }
    }

    int l16 = lane & 15, g = lane >> 4;
    #pragma unroll
    for (int mi = 0; mi < 2; ++mi) {
        #pragma unroll
        for (int j = 0; j < 4; ++j) {
            #pragma unroll
            for (int i = 0; i < 8; ++i) {
                int row = m0 + mi * 16 + i + g * 8;   // C/D layout: lane>=16 -> M += 8
                int col = n0 + j * 16 + l16;
                float v = acc[mi][j][i];
                if (bias)  v += bias[col];
                if (act)   v = gelu_exact(v);
                if (resid) v += resid[(size_t)row * N + col];
                if (Cf) Cf[(size_t)row * N + col] = v;
                if (Cb) Cb[(size_t)row * N + col] = (_Float16)v;
            }
        }
    }
}

// ---------------- flash attention (causal, online softmax) ----------------
// grid = (T/64, B*H), block = 128 (4 waves); each wave owns 16 query rows.
__global__ void __launch_bounds__(128)
flash_attn_kernel(const _Float16* __restrict__ qkv, const _Float16* __restrict__ vt,
                  _Float16* __restrict__ attnb) {
    __shared__ __align__(16) _Float16 plds[4][16][32];
    int lane = threadIdx.x & 31;
    int wave = threadIdx.x >> 5;
    int l16 = lane & 15, g = lane >> 4;
    int bh = blockIdx.y;
    int b = bh / NHEAD, h = bh % NHEAD;
    int q0 = blockIdx.x * 64 + wave * 16;

    const _Float16* Qb = qkv + ((size_t)b * SEQ) * QKVLD + h * HEADD;
    const _Float16* Kb = Qb + DMODEL;
    const _Float16* Vt = vt + (size_t)bh * HEADD * SEQ;   // [HEADD][SEQ]

    v16h aq0 = load_a_frag(Qb, QKVLD, q0, 0,  lane);
    v16h aq1 = load_a_frag(Qb, QKVLD, q0, 32, lane);

    v8f o[4];
    #pragma unroll
    for (int j = 0; j < 4; ++j)
        #pragma unroll
        for (int q = 0; q < 8; ++q) o[j][q] = 0.f;
    float mi[8], li[8];
    #pragma unroll
    for (int i = 0; i < 8; ++i) { mi[i] = -3e38f; li[i] = 0.f; }
    const float scale = 0.125f;   // 1/sqrt(64)

    for (int s0 = 0; s0 < q0 + 16; s0 += 32) {
        v8f sa0, sa1;
        #pragma unroll
        for (int q = 0; q < 8; ++q) { sa0[q] = 0.f; sa1[q] = 0.f; }
        // S = Q @ K^T over HD=64 (two k-chunks)
        sa0 = wmma_f16(aq0, load_b_frag(Kb, QKVLD, s0,      0,  lane), sa0);
        sa1 = wmma_f16(aq0, load_b_frag(Kb, QKVLD, s0 + 16, 0,  lane), sa1);
        sa0 = wmma_f16(aq1, load_b_frag(Kb, QKVLD, s0,      32, lane), sa0);
        sa1 = wmma_f16(aq1, load_b_frag(Kb, QKVLD, s0 + 16, 32, lane), sa1);

        #pragma unroll
        for (int i = 0; i < 8; ++i) {
            int qrow = q0 + i + 8 * g;
            int key0 = s0 + l16, key1 = s0 + 16 + l16;
            float v0 = (key0 <= qrow) ? sa0[i] * scale : -3e38f;
            float v1 = (key1 <= qrow) ? sa1[i] * scale : -3e38f;
            float rm = fmaxf(v0, v1);
            for (int msk = 1; msk < 16; msk <<= 1) rm = fmaxf(rm, __shfl_xor(rm, msk, 32));
            float mn = fmaxf(mi[i], rm);
            float al = __expf(mi[i] - mn);
            float p0 = (key0 <= qrow) ? __expf(v0 - mn) : 0.f;
            float p1 = (key1 <= qrow) ? __expf(v1 - mn) : 0.f;
            float rs = p0 + p1;
            for (int msk = 1; msk < 16; msk <<= 1) rs += __shfl_xor(rs, msk, 32);
            li[i] = li[i] * al + rs;
            mi[i] = mn;
            plds[wave][i + 8 * g][l16]      = (_Float16)p0;
            plds[wave][i + 8 * g][l16 + 16] = (_Float16)p1;
            #pragma unroll
            for (int j = 0; j < 4; ++j) o[j][i] *= al;
        }
        asm volatile("s_wait_dscnt 0" ::: "memory");   // wave-private LDS RAW
        v16h pa = load_a_frag(&plds[wave][0][0], 32, 0, 0, lane);
        #pragma unroll
        for (int j = 0; j < 4; ++j)
            o[j] = wmma_f16(pa, load_b_frag(Vt, SEQ, j * 16, s0, lane), o[j]);
    }

    #pragma unroll
    for (int i = 0; i < 8; ++i) {
        int qrow = q0 + i + 8 * g;
        float inv = 1.0f / li[i];
        #pragma unroll
        for (int j = 0; j < 4; ++j) {
            float val = o[j][i] * inv;
            attnb[((size_t)(b * SEQ) + qrow) * DMODEL + h * HEADD + j * 16 + l16] = (_Float16)val;
        }
    }
}

// ---------------- host orchestration ----------------
static inline size_t align256(size_t x) { return (x + 255) & ~(size_t)255; }

extern "C" void kernel_launch(void* const* d_in, const int* in_sizes, int n_in,
                              void* d_out, int out_size, void* d_ws, size_t ws_size,
                              hipStream_t stream) {
    (void)in_sizes; (void)n_in; (void)out_size; (void)ws_size;
    const int*   idx   = (const int*)  d_in[0];
    const float* tok   = (const float*)d_in[1];
    const float* pos   = (const float*)d_in[2];
    const float* ln1w  = (const float*)d_in[3];
    const float* ln1b  = (const float*)d_in[4];
    const float* Wqkv  = (const float*)d_in[5];
    const float* bqkv  = (const float*)d_in[6];
    const float* Wproj = (const float*)d_in[7];
    const float* bproj = (const float*)d_in[8];
    const float* ln2w  = (const float*)d_in[9];
    const float* ln2b  = (const float*)d_in[10];
    const float* Wfc   = (const float*)d_in[11];
    const float* bfc   = (const float*)d_in[12];
    const float* Wfc2  = (const float*)d_in[13];
    const float* bfc2  = (const float*)d_in[14];
    const float* lnfw  = (const float*)d_in[15];
    const float* lnfb  = (const float*)d_in[16];

    // workspace carve (all recomputed every call; deterministic)
    char* w = (char*)d_ws; size_t off = 0;
    auto carve = [&](size_t bytes) -> void* { void* p = w + off; off = align256(off + bytes); return p; };
    _Float16* tokh   = (_Float16*)carve((size_t)VOCAB * DMODEL * 2);
    float*    x      = (float*)   carve((size_t)BT * DMODEL * 4);
    _Float16* hb     = (_Float16*)carve((size_t)BT * DMODEL * 2);
    _Float16* qkvb   = (_Float16*)carve((size_t)BT * QKVLD * 2);
    _Float16* vth    = (_Float16*)carve((size_t)BATCH * NHEAD * HEADD * SEQ * 2);
    _Float16* attnb  = (_Float16*)carve((size_t)BT * DMODEL * 2);
    _Float16* ffb    = (_Float16*)carve((size_t)BT * FFDIM * 2);
    _Float16* wqkvT  = (_Float16*)carve((size_t)QKVLD * DMODEL * 2);
    _Float16* wprojT = (_Float16*)carve((size_t)DMODEL * DMODEL * 2);
    _Float16* wfcT   = (_Float16*)carve((size_t)FFDIM * DMODEL * 2);
    _Float16* wfc2T  = (_Float16*)carve((size_t)DMODEL * FFDIM * 2);

    auto nblk = [](long n) { return (int)((n + 255) / 256); };

    // tied embedding table in f16 (also Bt of the LM head: [V, D] row-major == [N,K])
    cvt_f16_kernel<<<nblk((long)VOCAB * DMODEL), 256, 0, stream>>>(tok, tokh, (long)VOCAB * DMODEL);
    // x = tok_emb[idx] + pos_emb
    embed_kernel<<<nblk((long)BT * DMODEL), 256, 0, stream>>>(idx, tok, pos, x);

    for (int l = 0; l < NLAYER; ++l) {
        // per-layer weight transposes (f32 [K,N] -> f16 [N,K])
        transpose_cvt_kernel<<<nblk((long)DMODEL * QKVLD), 256, 0, stream>>>(
            Wqkv + (size_t)l * DMODEL * QKVLD, wqkvT, DMODEL, QKVLD);
        transpose_cvt_kernel<<<nblk((long)DMODEL * DMODEL), 256, 0, stream>>>(
            Wproj + (size_t)l * DMODEL * DMODEL, wprojT, DMODEL, DMODEL);
        transpose_cvt_kernel<<<nblk((long)DMODEL * FFDIM), 256, 0, stream>>>(
            Wfc + (size_t)l * DMODEL * FFDIM, wfcT, DMODEL, FFDIM);
        transpose_cvt_kernel<<<nblk((long)FFDIM * DMODEL), 256, 0, stream>>>(
            Wfc2 + (size_t)l * FFDIM * DMODEL, wfc2T, FFDIM, DMODEL);

        // ln1 -> h (f16)
        layernorm_kernel<<<BT, 256, 0, stream>>>(x, ln1w + (size_t)l * DMODEL, ln1b + (size_t)l * DMODEL, hb);
        // qkv = h @ Wqkv + bqkv  (f16 out)
        gemm_f16_kernel<<<dim3(QKVLD / 128, BT / 64), 128, 0, stream>>>(
            hb, wqkvT, bqkv + (size_t)l * QKVLD, nullptr, nullptr, qkvb, BT, QKVLD, DMODEL, 0);
        // V transpose per head
        vtrans_kernel<<<nblk((long)BT * DMODEL), 256, 0, stream>>>(qkvb, vth);
        // attention -> attnb (f16)
        flash_attn_kernel<<<dim3(SEQ / 64, BATCH * NHEAD), 128, 0, stream>>>(qkvb, vth, attnb);
        // x = x + attnb @ Wproj + bproj
        gemm_f16_kernel<<<dim3(DMODEL / 128, BT / 64), 128, 0, stream>>>(
            attnb, wprojT, bproj + (size_t)l * DMODEL, x, x, nullptr, BT, DMODEL, DMODEL, 0);
        // ln2 -> h
        layernorm_kernel<<<BT, 256, 0, stream>>>(x, ln2w + (size_t)l * DMODEL, ln2b + (size_t)l * DMODEL, hb);
        // ff = gelu(h @ Wfc + bfc)  (f16 out)
        gemm_f16_kernel<<<dim3(FFDIM / 128, BT / 64), 128, 0, stream>>>(
            hb, wfcT, bfc + (size_t)l * FFDIM, nullptr, nullptr, ffb, BT, FFDIM, DMODEL, 1);
        // x = x + ff @ Wfc2 + bfc2
        gemm_f16_kernel<<<dim3(DMODEL / 128, BT / 64), 128, 0, stream>>>(
            ffb, wfc2T, bfc2 + (size_t)l * DMODEL, x, x, nullptr, BT, DMODEL, FFDIM, 0);
    }

    // final LN + tied LM head: logits = h @ tok_emb^T  (f32 -> d_out)
    layernorm_kernel<<<BT, 256, 0, stream>>>(x, lnfw, lnfb, hb);
    gemm_f16_kernel<<<dim3(VOCAB / 128, BT / 64), 128, 0, stream>>>(
        hb, tokh, nullptr, nullptr, (float*)d_out, nullptr, BT, VOCAB, DMODEL, 0);
}
